// ProbAttention_32856499814997
// MI455X (gfx1250) — matrix operand order repaired and checked by reading, not compiled, not run on hardware
//
#include <hip/hip_runtime.h>
#include <hip/hip_bf16.h>
#include <math.h>

// ---------------- problem constants (from reference) ----------------
#define BB    4
#define LL    2048
#define DD    512
#define HH    8
#define DKK   64
#define SKK   40      // 5 * ceil(ln(2048))
#define NTOPK 40

#define MROWS (BB * LL)          // 8192
#define NEG_INF (-3.402823466e+38f)

typedef _Float16 v16h __attribute__((ext_vector_type(16)));
typedef _Float16 v8h  __attribute__((ext_vector_type(8)));
typedef _Float16 v4h  __attribute__((ext_vector_type(4)));
typedef float    v8f  __attribute__((ext_vector_type(8)));
typedef int      v4i  __attribute__((ext_vector_type(4)));

union v16h_u { v16h v; v8h h[2]; };

// ---------------- CDNA5 async global->LDS path (probe-guarded) -------
#if __has_builtin(__builtin_amdgcn_global_load_async_to_lds_b128)
#define HAVE_ASYNC_LDS 1
#else
#define HAVE_ASYNC_LDS 0
#endif

#if __has_builtin(__builtin_amdgcn_s_wait_asynccnt)
#define WAIT_ASYNC(n) __builtin_amdgcn_s_wait_asynccnt(n)
#elif HAVE_ASYNC_LDS
#define WAIT_ASYNC(n) asm volatile("s_wait_asynccnt %0" ::"i"(n))
#else
#define WAIT_ASYNC(n) ((void)0)
#endif

__device__ __forceinline__ void async_copy16(const _Float16* g, _Float16* l) {
#if HAVE_ASYNC_LDS
    // signature (from hipcc diagnostic): (AS1 v4i* src, AS3 v4i* dst, imm off, imm cpol)
    __builtin_amdgcn_global_load_async_to_lds_b128(
        (__attribute__((address_space(1))) v4i*)(void*)g,
        (__attribute__((address_space(3))) v4i*)(void*)l, 0, 0);
#else
    *(v8h*)l = *(const v8h*)g;   // sync fallback: global->VGPR->LDS
#endif
}

// =====================================================================
// conversion kernels (run once; amortized over 8x L2 reuse in the GEMMs)
// =====================================================================
__global__ __launch_bounds__(256)
void k_cvt_f16(const float* __restrict__ src, _Float16* __restrict__ dst) {
    const int i = blockIdx.x * 256 + threadIdx.x;   // element-quad index
    const float4 f = ((const float4*)src)[i];
    v4h o;
    o[0] = (_Float16)f.x; o[1] = (_Float16)f.y;
    o[2] = (_Float16)f.z; o[3] = (_Float16)f.w;
    ((v4h*)dst)[i] = o;
}

// Wt[n][k] = (f16) W[k][n]   (512x512, LDS tiled transpose)
__global__ __launch_bounds__(256)
void k_cvt_w_t(const float* __restrict__ W, _Float16* __restrict__ Wt) {
    __shared__ _Float16 tile[32][33];
    const int bn = blockIdx.x * 32;     // n range
    const int bk = blockIdx.y * 32;     // k range
    const int tx = threadIdx.x & 31;
    const int ty = threadIdx.x >> 5;    // 0..7
    #pragma unroll
    for (int i = 0; i < 4; ++i) {
        const int kr = ty + i * 8;
        tile[kr][tx] = (_Float16)W[(size_t)(bk + kr) * DD + bn + tx];
    }
    __syncthreads();
    #pragma unroll
    for (int i = 0; i < 4; ++i) {
        const int nr = ty + i * 8;
        Wt[(size_t)(bn + nr) * DD + bk + tx] = tile[tx][nr];
    }
}

// =====================================================================
// Kernel 1: Y[M,N] = Xh[M,K] @ Wt^T + bias    (M=8192, K=N=512, f16 ops)
// Double-buffered LDS, async global->LDS staging, wmma_f32_16x16x32_f16.
// Block: 256 threads = 8 waves; tile 128(M) x 64(N); BK=32.
// Per stage each wave issues 3 async b128 copies -> s_wait_asynccnt 3
// overlaps next-stage DMA with current-stage WMMAs.
// =====================================================================
#define LDA 40
#define LDB 40

template <typename TOUT>
__global__ __launch_bounds__(256)
void proj_wmma(const _Float16* __restrict__ Xh, const _Float16* __restrict__ Wt,
               const float* __restrict__ bias, TOUT* __restrict__ Y) {
    __shared__ __align__(16) _Float16 As[2][128 * LDA];
    __shared__ __align__(16) _Float16 Bs[2][64 * LDB];

    const int colBase = blockIdx.x * 64;
    const int rowBase = blockIdx.y * 128;
    const int tid  = threadIdx.x;
    const int lane = tid & 31;
    const int wave = tid >> 5;
    const int m    = lane & 15;
    const int hi   = lane >> 4;

    // this thread's staging chunks (16B each): 2 for A (512 chunks), 1 for B (256)
    const int ar0 = tid >> 2,          ak0 = (tid & 3) * 8;
    const int ar1 = (tid + 256) >> 2,  ak1 = (tid & 3) * 8;   // +256 keeps (c&3)
    const int bn  = tid >> 2,          bk  = (tid & 3) * 8;

    v8f acc[4] = {v8f{}, v8f{}, v8f{}, v8f{}};

    // stage K-step kk into LDS buffer `buf`
    auto stage = [&](int buf, int kk) {
        async_copy16(&Xh[(size_t)(rowBase + ar0) * DD + kk + ak0], &As[buf][ar0 * LDA + ak0]);
        async_copy16(&Xh[(size_t)(rowBase + ar1) * DD + kk + ak1], &As[buf][ar1 * LDA + ak1]);
        async_copy16(&Wt[(size_t)(colBase + bn)  * DD + kk + bk ], &Bs[buf][bn  * LDB + bk ]);
    };

    const int NSTEP = DD / 32;   // 16
    stage(0, 0);
    for (int step = 0; step < NSTEP; ++step) {
        const int cur = step & 1;
        if (step + 1 < NSTEP) {
            stage(cur ^ 1, (step + 1) * 32);
            WAIT_ASYNC(3);       // current buffer's 3 copies done; next's in flight
        } else {
            WAIT_ASYNC(0);
        }
        __syncthreads();

        // A fragment (16x32 f16): lanes 0-15 row m K{0..7,16..23},
        // lanes 16-31 row m K{8..15,24..31}
        const _Float16* arow = &As[cur][(wave * 16 + m) * LDA];
        v16h_u a;
        a.h[0] = *(const v8h*)&arow[hi * 8];
        a.h[1] = *(const v8h*)&arow[16 + hi * 8];

        #pragma unroll
        for (int n = 0; n < 4; ++n) {
            const _Float16* brow = &Bs[cur][(n * 16 + m) * LDB + hi * 8];
            v16h_u bb;
            bb.h[0] = *(const v8h*)&brow[0];
            bb.h[1] = *(const v8h*)&brow[16];
            acc[n] = __builtin_amdgcn_wmma_f32_16x16x32_f16(
                false, a.v, false, bb.v, (short)0, acc[n], false, false);
        }
        __syncthreads();   // all waves done reading `cur` before it is re-staged
    }

    // epilogue: bias + store. C layout: VGPR r -> row (hi?8+r:r), lane m -> col
    #pragma unroll
    for (int n = 0; n < 4; ++n) {
        const int col = colBase + n * 16 + m;
        const float bv = bias[col];
        #pragma unroll
        for (int r = 0; r < 8; ++r) {
            const int row = rowBase + wave * 16 + hi * 8 + r;
            Y[(size_t)row * DD + col] = (TOUT)(acc[n][r] + bv);
        }
    }
}

// =====================================================================
// Kernel 2: sparsity measure  M[b,h,q] = max_j(QK) - sum_j(QK)/L
// =====================================================================
__global__ __launch_bounds__(256)
void k_measure(const _Float16* __restrict__ Qh, const _Float16* __restrict__ Kh,
               const int* __restrict__ idx, float* __restrict__ M) {
    const int t = blockIdx.x * 256 + threadIdx.x;     // 0..B*H*L-1
    const int q = t % LL;
    const int h = (t / LL) % HH;
    const int b = t / (LL * HH);
    const _Float16* qrow = Qh + ((size_t)b * LL + q) * DD + h * DKK;
    float qreg[DKK];
    #pragma unroll
    for (int k = 0; k < DKK; ++k) qreg[k] = (float)qrow[k];
    float mx = NEG_INF, sum = 0.f;
    for (int j = 0; j < SKK; ++j) {
        const int s = idx[q * SKK + j];
        const _Float16* krow = Kh + ((size_t)b * LL + s) * DD + h * DKK;
        float d = 0.f;
        #pragma unroll
        for (int k = 0; k < DKK; ++k) d += qreg[k] * (float)krow[k];
        mx = fmaxf(mx, d);
        sum += d;
    }
    M[t] = mx - sum * (1.0f / (float)LL);
}

// =====================================================================
// Kernel 3: top-NTOP indices per (b,h) by iterative block argmax
// =====================================================================
__global__ __launch_bounds__(256)
void k_topk(const float* __restrict__ M, int* __restrict__ Mtop) {
    const int bh = blockIdx.x;
    __shared__ float vals[LL];
    __shared__ float rv[256];
    __shared__ int   ri[256];
    const float* Mrow = M + (size_t)bh * LL;
    for (int i = threadIdx.x; i < LL; i += 256) vals[i] = Mrow[i];
    __syncthreads();
    for (int it = 0; it < NTOPK; ++it) {
        float best = NEG_INF; int bi = 0;
        for (int i = threadIdx.x; i < LL; i += 256) {
            const float v = vals[i];
            if (v > best) { best = v; bi = i; }
        }
        rv[threadIdx.x] = best; ri[threadIdx.x] = bi;
        __syncthreads();
        for (int s = 128; s > 0; s >>= 1) {
            if (threadIdx.x < s) {
                const float vo = rv[threadIdx.x + s];
                const int   io = ri[threadIdx.x + s];
                if (vo > rv[threadIdx.x] ||
                    (vo == rv[threadIdx.x] && io < ri[threadIdx.x])) {
                    rv[threadIdx.x] = vo; ri[threadIdx.x] = io;
                }
            }
            __syncthreads();
        }
        if (threadIdx.x == 0) {
            Mtop[bh * NTOPK + it] = ri[0];
            vals[ri[0]] = NEG_INF;
        }
        __syncthreads();
    }
}

// =====================================================================
// Kernel 4: Vsum[b,h,k] = sum_s V[b,s,h,k]
// =====================================================================
__global__ __launch_bounds__(256)
void k_vsum(const _Float16* __restrict__ Vh, float* __restrict__ Vsum) {
    const int bh = blockIdx.x;
    const int b = bh / HH, h = bh % HH;
    const int col = threadIdx.x & 63;
    const int rg  = threadIdx.x >> 6;
    float s = 0.f;
    for (int i = rg * (LL / 4); i < (rg + 1) * (LL / 4); ++i)
        s += (float)Vh[((size_t)b * LL + i) * DD + h * DKK + col];
    __shared__ float red[256];
    red[threadIdx.x] = s;
    __syncthreads();
    if (rg == 0)
        Vsum[bh * DKK + col] = s + red[64 + col] + red[128 + col] + red[192 + col];
}

// =====================================================================
// Kernel 5: ctx broadcast init (f16)
// =====================================================================
__global__ __launch_bounds__(256)
void k_ctx_init(const float* __restrict__ Vsum, _Float16* __restrict__ ctx) {
    const size_t i = (size_t)blockIdx.x * 256 + threadIdx.x;
    if (i >= (size_t)BB * LL * DD) return;
    const int d = (int)(i % DD);
    const int b = (int)(i / ((size_t)LL * DD));
    ctx[i] = (_Float16)Vsum[(b * HH + (d / DKK)) * DKK + (d % DKK)];
}

// =====================================================================
// Kernel 6: attention for the NTOP selected queries, scatter into ctx
// =====================================================================
__global__ __launch_bounds__(256)
void k_attn(const _Float16* __restrict__ Qh, const _Float16* __restrict__ Kh,
            const _Float16* __restrict__ Vh, const int* __restrict__ Mtop,
            _Float16* __restrict__ ctx) {
    const int ub = blockIdx.x;
    const int u = ub % NTOPK;
    const int h = (ub / NTOPK) % HH;
    const int b = ub / (NTOPK * HH);
    const int q = Mtop[(b * HH + h) * NTOPK + u];

    __shared__ float qs[DKK];
    __shared__ float sc[LL];
    __shared__ float red[256];

    if (threadIdx.x < DKK)
        qs[threadIdx.x] = (float)Qh[((size_t)b * LL + q) * DD + h * DKK + threadIdx.x];
    __syncthreads();

    const float scale = 0.125f;   // 1/sqrt(64)
    float lmax = NEG_INF;
    for (int s = threadIdx.x; s < LL; s += 256) {
        const _Float16* krow = Kh + ((size_t)b * LL + s) * DD + h * DKK;
        float d = 0.f;
        #pragma unroll
        for (int k = 0; k < DKK; ++k) d += qs[k] * (float)krow[k];
        d *= scale;
        sc[s] = d;
        lmax = fmaxf(lmax, d);
    }
    red[threadIdx.x] = lmax;
    __syncthreads();
    for (int s = 128; s > 0; s >>= 1) {
        if (threadIdx.x < s)
            red[threadIdx.x] = fmaxf(red[threadIdx.x], red[threadIdx.x + s]);
        __syncthreads();
    }
    const float mrow = red[0];
    __syncthreads();

    float lsum = 0.f;
    for (int s = threadIdx.x; s < LL; s += 256) {
        const float e = __expf(sc[s] - mrow);
        sc[s] = e;
        lsum += e;
    }
    red[threadIdx.x] = lsum;
    __syncthreads();
    for (int s = 128; s > 0; s >>= 1) {
        if (threadIdx.x < s)
            red[threadIdx.x] += red[threadIdx.x + s];
        __syncthreads();
    }
    const float invZ = 1.0f / red[0];
    __syncthreads();

    const int col = threadIdx.x & 63;
    const int rg  = threadIdx.x >> 6;
    float acc = 0.f;
    for (int s = rg * (LL / 4); s < (rg + 1) * (LL / 4); ++s)
        acc += sc[s] * (float)Vh[((size_t)b * LL + s) * DD + h * DKK + col];
    red[threadIdx.x] = acc;
    __syncthreads();
    if (rg == 0) {
        const float tot = acc + red[64 + col] + red[128 + col] + red[192 + col];
        ctx[((size_t)b * LL + q) * DD + h * DKK + col] = (_Float16)(tot * invZ);
    }
}

// =====================================================================
// Launcher
// =====================================================================
extern "C" void kernel_launch(void* const* d_in, const int* in_sizes, int n_in,
                              void* d_out, int out_size, void* d_ws, size_t ws_size,
                              hipStream_t stream) {
    (void)in_sizes; (void)n_in; (void)out_size; (void)ws_size;

    const float* queries = (const float*)d_in[0];
    const float* keys    = (const float*)d_in[1];
    const float* values  = (const float*)d_in[2];
    // d_in[3] = attn_mask (unused by reference math)
    const int*   idx     = (const int*)d_in[4];
    const float* Wq = (const float*)d_in[5];
    const float* bq = (const float*)d_in[6];
    const float* Wk = (const float*)d_in[7];
    const float* bk = (const float*)d_in[8];
    const float* Wv = (const float*)d_in[9];
    const float* bv = (const float*)d_in[10];
    const float* Wo = (const float*)d_in[11];
    const float* bo = (const float*)d_in[12];
    float* out = (float*)d_out;

    // ---- workspace carve-up (256B-aligned) ----
    char* p = (char*)d_ws;
    auto alloc = [&](size_t bytes) {
        char* r = p;
        p += (bytes + 255) & ~(size_t)255;
        return r;
    };
    const size_t TEN = (size_t)BB * LL * DD;        // 4,194,304 elements
    _Float16* Xq  = (_Float16*)alloc(TEN * 2);      // f16 converted inputs
    _Float16* Xk  = (_Float16*)alloc(TEN * 2);
    _Float16* Xv  = (_Float16*)alloc(TEN * 2);
    _Float16* Wqt = (_Float16*)alloc((size_t)DD * DD * 2);  // f16 transposed weights
    _Float16* Wkt = (_Float16*)alloc((size_t)DD * DD * 2);
    _Float16* Wvt = (_Float16*)alloc((size_t)DD * DD * 2);
    _Float16* Wot = (_Float16*)alloc((size_t)DD * DD * 2);
    _Float16* Qh  = (_Float16*)alloc(TEN * 2);      // projected Q/K/V (f16)
    _Float16* Kh  = (_Float16*)alloc(TEN * 2);
    _Float16* Vh  = (_Float16*)alloc(TEN * 2);
    _Float16* ctx = (_Float16*)alloc(TEN * 2);
    float*    Mm  = (float*)alloc((size_t)BB * HH * LL * 4);
    float*    Vs  = (float*)alloc((size_t)BB * HH * DKK * 4);
    int*      Mtp = (int*)alloc((size_t)BB * HH * NTOPK * 4);

    const dim3 blk(256);
    const dim3 gcvt((unsigned)(TEN / 4 / 256));     // 4096 blocks
    const dim3 gwt(DD / 32, DD / 32);               // (16,16)
    const dim3 gproj(DD / 64, MROWS / 128);         // (8,64)

    // 0: one-time f16 conversions (activations + transposed weights)
    k_cvt_f16<<<gcvt, blk, 0, stream>>>(queries, Xq);
    k_cvt_f16<<<gcvt, blk, 0, stream>>>(keys,    Xk);
    k_cvt_f16<<<gcvt, blk, 0, stream>>>(values,  Xv);
    k_cvt_w_t<<<gwt, blk, 0, stream>>>(Wq, Wqt);
    k_cvt_w_t<<<gwt, blk, 0, stream>>>(Wk, Wkt);
    k_cvt_w_t<<<gwt, blk, 0, stream>>>(Wv, Wvt);
    k_cvt_w_t<<<gwt, blk, 0, stream>>>(Wo, Wot);

    // 1-3: input projections (async-staged WMMA GEMMs, f16 out)
    proj_wmma<_Float16><<<gproj, blk, 0, stream>>>(Xq, Wqt, bq, Qh);
    proj_wmma<_Float16><<<gproj, blk, 0, stream>>>(Xk, Wkt, bk, Kh);
    proj_wmma<_Float16><<<gproj, blk, 0, stream>>>(Xv, Wvt, bv, Vh);

    // 4: sparsity measure
    k_measure<<<dim3((BB * HH * LL) / 256), blk, 0, stream>>>(Qh, Kh, idx, Mm);

    // 5: top-k query selection
    k_topk<<<dim3(BB * HH), blk, 0, stream>>>(Mm, Mtp);

    // 6-7: V column sums, broadcast into ctx
    k_vsum<<<dim3(BB * HH), blk, 0, stream>>>(Vh, Vs);
    k_ctx_init<<<dim3((unsigned)((TEN + 255) / 256)), blk, 0, stream>>>(Vs, ctx);

    // 8: attention for selected queries, scattered into ctx
    k_attn<<<dim3(BB * HH * NTOPK), blk, 0, stream>>>(Qh, Kh, Vh, Mtp, ctx);

    // 9: output projection (WMMA GEMM, f32 out) -> d_out
    proj_wmma<float><<<gproj, blk, 0, stream>>>(ctx, Wot, bo, out);
}
